// InverseSpectralProjection_25486335934755
// MI455X (gfx1250) — matrix operand order repaired and checked by reading, not compiled, not run on hardware
//
#include <hip/hip_runtime.h>
#include <hip/hip_bf16.h>

// out = scale(n,m) * (A · z · A^T)  per (b,c) image, where
//   A[n,h]      = cos(pi * n * (2h+1) / 512)   (256x256 DCT-II matrix)
//   scale(n,m)  = cos(pi*(n+m)/512) / 65536
// Mathematically identical to flip-extend + ifft2 + crop in the reference.

typedef __attribute__((ext_vector_type(16))) _Float16 v16h;
typedef __attribute__((ext_vector_type(8)))  float    v8f;

#define PI_OVER_512 0.0061359231515425649f

// ---------------------------------------------------------------------------
// Build the split-precision DCT matrix: Ah + Al ≈ cos(pi*n*(2h+1)/512), f32-
// accurate to ~2^-22. Integer reduction mod 1024 keeps the cos argument small
// so f32 cosf range reduction is exact.
// ---------------------------------------------------------------------------
__global__ void isp_build_dct(_Float16* __restrict__ Ah, _Float16* __restrict__ Al) {
    int idx = blockIdx.x * 256 + threadIdx.x;   // 65536 elements
    int n = idx >> 8;
    int h = idx & 255;
    int k = (n * (2 * h + 1)) & 1023;           // cos(pi*k/512) has period 1024 in k
    float c = cosf((float)k * PI_OVER_512);
    _Float16 hi = (_Float16)c;
    Ah[idx] = hi;
    Al[idx] = (_Float16)(c - (float)hi);
}

// ---------------------------------------------------------------------------
// Fragment index helpers for wave32 WMMA 16x16x32 f16 (CDNA5 ISA 7.12.2):
//  A (16x32, MxK): lane half 0 -> K {0..7,16..23}, half 1 -> K {8..15,24..31}
//  B (32x16, KxN): lane half 0 -> K {0..7,16..23}, half 1 -> K {8..15,24..31}
//  C (16x16 f32): VGPR r, lane l -> M = r + 8*(l>>4), N = l&15
// ---------------------------------------------------------------------------
__device__ __forceinline__ int ka_idx(int e, int half) {
    return (e & 7) | (half << 3) | ((e >> 3) << 4);
}
__device__ __forceinline__ int kb_idx(int e, int half) {
    return (e & 7) | (half << 3) | ((e >> 3) << 4);
}

__device__ __forceinline__ v8f wmma_f16(v16h a, v16h b, v8f c) {
    return __builtin_amdgcn_wmma_f32_16x16x32_f16(false, a, false, b, (short)0, c,
                                                  false, false);
}

// ---------------------------------------------------------------------------
// One workgroup = 16 waves = one 16-row output strip of one 256x256 image.
// Phase 1: strip T[0:16, 0:256] = A[n0:n0+16, :] * z   -> LDS (split f16)
// Phase 2: out[n0:n0+16, :]     = scale ⊙ (T * A^T)
// ---------------------------------------------------------------------------
__global__ void __launch_bounds__(512)
isp_dct2d(const float* __restrict__ z,
          const _Float16* __restrict__ Ah, const _Float16* __restrict__ Al,
          float* __restrict__ out) {
    __shared__ _Float16 Sh[16 * 256];   // 8 KB: T strip, hi halves
    __shared__ _Float16 Sl[16 * 256];   // 8 KB: T strip, lo halves

    const int img  = blockIdx.x >> 4;            // 0..255  (b*32 + c)
    const int n0   = (blockIdx.x & 15) << 4;     // output-row tile base
    const int wave = threadIdx.x >> 5;           // 0..15
    const int lane = threadIdx.x & 31;
    const int half = lane >> 4;
    const int lrow = lane & 15;

    const float* zimg = z + (size_t)img * 65536;

    // ---------------- Phase 1: T tile at columns [wave*16, wave*16+16) -----
    {
        const int w0 = wave << 4;
        v8f acc = {};
        const int arow = (n0 + lrow) * 256;      // A row for this lane
        const int bcol = w0 + lrow;              // z column for this lane
        for (int k0 = 0; k0 < 256; k0 += 32) {
            v16h ah, al, bh, bl;
#pragma unroll
            for (int e = 0; e < 16; ++e) {
                int ai = arow + k0 + ka_idx(e, half);
                ah[e] = Ah[ai];
                al[e] = Al[ai];
            }
#pragma unroll
            for (int e = 0; e < 16; ++e) {
                int kk = k0 + kb_idx(e, half);
                float v = zimg[kk * 256 + bcol]; // split z on the fly
                _Float16 hv = (_Float16)v;
                bh[e] = hv;
                bl[e] = (_Float16)(v - (float)hv);
            }
            acc = wmma_f16(ah, bh, acc);         // hi*hi
            acc = wmma_f16(ah, bl, acc);         // hi*lo
            acc = wmma_f16(al, bh, acc);         // lo*hi
        }
        // Scatter C tile into the LDS strip, split hi/lo.
#pragma unroll
        for (int r = 0; r < 8; ++r) {
            int nl = r + (half << 3);            // local strip row 0..15
            int w  = w0 + lrow;
            float v = acc[r];
            _Float16 hv = (_Float16)v;
            Sh[nl * 256 + w] = hv;
            Sl[nl * 256 + w] = (_Float16)(v - (float)hv);
        }
    }
    __syncthreads();

    // ---------------- Phase 2: out tile at columns [wave*16, wave*16+16) ---
    {
        const int m0 = wave << 4;
        v8f acc = {};
        const int trow = lrow * 256;             // T strip row for this lane
        const int brow = (m0 + lrow) * 256;      // A row m (B = A^T -> A[m, w])
        for (int k0 = 0; k0 < 256; k0 += 32) {
            v16h th, tl, bh, bl;
#pragma unroll
            for (int e = 0; e < 16; ++e) {
                int ti = trow + k0 + ka_idx(e, half);
                th[e] = Sh[ti];
                tl[e] = Sl[ti];
            }
#pragma unroll
            for (int e = 0; e < 16; ++e) {
                int ai = brow + k0 + kb_idx(e, half);
                bh[e] = Ah[ai];
                bl[e] = Al[ai];
            }
            acc = wmma_f16(th, bh, acc);
            acc = wmma_f16(th, bl, acc);
            acc = wmma_f16(tl, bh, acc);
        }
        float* outimg = out + (size_t)img * 65536;
#pragma unroll
        for (int r = 0; r < 8; ++r) {
            int n = n0 + r + (half << 3);
            int m = m0 + lrow;
            float s = __cosf((float)(n + m) * PI_OVER_512) * (1.0f / 65536.0f);
            outimg[n * 256 + m] = acc[r] * s;
        }
    }
}

extern "C" void kernel_launch(void* const* d_in, const int* in_sizes, int n_in,
                              void* d_out, int out_size, void* d_ws, size_t ws_size,
                              hipStream_t stream) {
    (void)in_sizes; (void)n_in; (void)out_size; (void)ws_size;
    const float* z = (const float*)d_in[0];      // (8,32,256,256) f32
    float* out = (float*)d_out;                  // (8,32,256,256) f32

    // Workspace: split-precision DCT matrix, 2 * 65536 f16 = 256 KB.
    _Float16* Ah = (_Float16*)d_ws;
    _Float16* Al = Ah + 65536;

    isp_build_dct<<<256, 256, 0, stream>>>(Ah, Al);
    // 256 images * 16 row-strips; 512 threads (16 waves) per block.
    isp_dct2d<<<4096, 512, 0, stream>>>(z, Ah, Al, out);
}
